// DGCNNSemSeg_68212670595319
// MI455X (gfx1250) — matrix operand order repaired and checked by reading, not compiled, not run on hardware
//
#include <hip/hip_runtime.h>

#define NPTS   4096
#define NBATCH 4
#define KNN    20
#define SEDGE  (NPTS * KNN)   // 81920
#define SCHUNK 64

typedef _Float16 half_t;
typedef __attribute__((ext_vector_type(16))) _Float16 v16h;
typedef __attribute__((ext_vector_type(4)))  _Float16 v4h;
typedef __attribute__((ext_vector_type(8)))  float    v8f;
typedef __attribute__((ext_vector_type(4)))  float    v4f;

// ---------------------------------------------------------------------------
// kNN stage 1: coords = feature channels 6..7, feature tensor is [b][c][n].
// ---------------------------------------------------------------------------
__global__ void knn2_kernel(const float* __restrict__ feat, int* __restrict__ idx) {
  int t = blockIdx.x * blockDim.x + threadIdx.x;
  if (t >= NBATCH * NPTS) return;
  int n = t % NPTS, b = t / NPTS;
  const float* c0 = feat + ((size_t)b * 8 + 6) * NPTS;
  const float* c1 = feat + ((size_t)b * 8 + 7) * NPTS;
  float q0 = c0[n], q1 = c1[n];
  float bd[KNN]; int bi[KNN];
#pragma unroll
  for (int j = 0; j < KNN; ++j) { bd[j] = 3.4e38f; bi[j] = 0; }
  for (int m = 0; m < NPTS; ++m) {
    float d0 = c0[m] - q0, d1 = c1[m] - q1;
    float dist = fmaf(d0, d0, d1 * d1);
    if (dist < bd[KNN - 1]) {
      int j = KNN - 1;
      while (j > 0 && bd[j - 1] > dist) { bd[j] = bd[j - 1]; bi[j] = bi[j - 1]; --j; }
      bd[j] = dist; bi[j] = m;
    }
  }
  int* op = idx + (size_t)t * KNN;
  for (int j = 0; j < KNN; ++j) op[j] = bi[j];
}

// kNN stages 2/3: x is point-major [b][n][64]
__global__ void knn64_kernel(const float* __restrict__ x, int* __restrict__ idx) {
  int t = blockIdx.x * blockDim.x + threadIdx.x;
  if (t >= NBATCH * NPTS) return;
  int n = t % NPTS, b = t / NPTS;
  const float* xb = x + (size_t)b * NPTS * 64;
  float q[64];
#pragma unroll
  for (int d = 0; d < 64; ++d) q[d] = xb[(size_t)n * 64 + d];
  float bd[KNN]; int bi[KNN];
#pragma unroll
  for (int j = 0; j < KNN; ++j) { bd[j] = 3.4e38f; bi[j] = 0; }
  for (int m = 0; m < NPTS; ++m) {
    const float* cm = xb + (size_t)m * 64;
    float dist = 0.f;
#pragma unroll
    for (int d = 0; d < 64; ++d) { float df = cm[d] - q[d]; dist = fmaf(df, df, dist); }
    if (dist < bd[KNN - 1]) {
      int j = KNN - 1;
      while (j > 0 && bd[j - 1] > dist) { bd[j] = bd[j - 1]; bi[j] = bi[j - 1]; --j; }
      bd[j] = dist; bi[j] = m;
    }
  }
  int* op = idx + (size_t)t * KNN;
  for (int j = 0; j < KNN; ++j) op[j] = bi[j];
}

// ---------------------------------------------------------------------------
// Stage-1 point GEMMs (C=8): G = W1[:, :8]*x, H = (W1[:,8:]-W1[:,:8])*x.
// Outputs point-major f32 [b][n][64].
// ---------------------------------------------------------------------------
__global__ void gh8_kernel(const float* __restrict__ w1, const float* __restrict__ feat,
                           float* __restrict__ G, float* __restrict__ H) {
  int t = blockIdx.x * blockDim.x + threadIdx.x;
  if (t >= NBATCH * NPTS) return;
  int n = t % NPTS, b = t / NPTS;
  float xv[8];
#pragma unroll
  for (int c = 0; c < 8; ++c) xv[c] = feat[((size_t)b * 8 + c) * NPTS + n];
  float* Gp = G + (size_t)t * 64;
  float* Hp = H + (size_t)t * 64;
  for (int o = 0; o < 64; ++o) {
    float ga = 0.f, ha = 0.f;
#pragma unroll
    for (int c = 0; c < 8; ++c) {
      float wg = w1[o * 16 + c];
      float wh = w1[o * 16 + 8 + c] - wg;
      ga = fmaf(wg, xv[c], ga);
      ha = fmaf(wh, xv[c], ha);
    }
    Gp[o] = ga; Hp[o] = ha;
  }
}

// ---------------------------------------------------------------------------
// Pack weights into per-lane WMMA A-fragment layout (f16).
// mode 0: W[o, koff+k]; mode 1: W[o, k+Cdiff] - W[o, k].
// ---------------------------------------------------------------------------
__global__ void pack_a_kernel(const float* __restrict__ W, int ldw, int koff, int Cdiff, int mode,
                              int O, int C, half_t* __restrict__ apack) {
  int t = blockIdx.x * blockDim.x + threadIdx.x;
  int nK = C >> 5;
  int total = (O >> 4) * nK * 32;
  if (t >= total) return;
  int lane = t & 31;
  int kt = (t >> 5) % nK;
  int mt = (t >> 5) / nK;
  int m  = (mt << 4) + (lane & 15);
  int ka = (kt << 5) + ((lane & 16) ? 16 : 0);
  half_t* dst = apack + (size_t)t * 16;
#pragma unroll
  for (int j = 0; j < 8; ++j) {
    int k = ka + 2 * j;
    float f0, f1;
    if (mode == 1) {
      f0 = W[(size_t)m * ldw + k + Cdiff]     - W[(size_t)m * ldw + k];
      f1 = W[(size_t)m * ldw + k + 1 + Cdiff] - W[(size_t)m * ldw + k + 1];
    } else {
      f0 = W[(size_t)m * ldw + koff + k];
      f1 = W[(size_t)m * ldw + koff + k + 1];
    }
    dst[2 * j]     = (half_t)f0;
    dst[2 * j + 1] = (half_t)f1;
  }
}

// ---------------------------------------------------------------------------
// WMMA GEMM. B is pre-activated f16, point-major with row stride ldb and
// channel offset coff: B(c,s) = bin[(b*S+s)*ldb + coff + c].
// out (f32, point-major [b][s][O]) = A(O,C) x B (+ bias[b][o]).
// One wave computes 16(M) x 64(S); K stepped by 32. C %32==0, S %64==0.
// Inner loop: 2 + 8 aligned b128 loads + 4 v_wmma, no conversions.
// ---------------------------------------------------------------------------
__global__ __launch_bounds__(256) void gemm_wmma_kernel(
    const half_t* __restrict__ apack, const half_t* __restrict__ bin,
    int ldb, int coff, int O, int C, int S,
    const float* __restrict__ bias, float* __restrict__ out) {
  int b    = blockIdx.y;
  int wid  = (blockIdx.x * blockDim.x + threadIdx.x) >> 5;
  int lane = threadIdx.x & 31;
  int nM = O >> 4, nS = S >> 6, nK = C >> 5;
  if (wid >= nM * nS) return;                 // wave-uniform guard: EXEC all ones
  int mt = wid % nM;
  int s0 = (wid / nM) << 6;
  int col = lane & 15;
  int kb  = (lane & 16) ? 16 : 0;

  const half_t* ap    = apack + ((size_t)mt * nK * 32 + lane) * 16;
  const half_t* bbase = bin + (size_t)b * S * ldb + coff;

  v8f acc[4] = {};
  for (int kt = 0; kt < nK; ++kt) {
    v16h a = *(const v16h*)ap;
    ap += 512;                                 // next K-tile (32 lanes * 16 halfs)
    int c0 = (kt << 5) + kb;
    v16h bf[4];
#pragma unroll
    for (int t = 0; t < 4; ++t) {
      int s = s0 + t * 16 + col;
      bf[t] = *(const v16h*)(bbase + (size_t)s * ldb + c0);
    }
#pragma unroll
    for (int t = 0; t < 4; ++t)
      acc[t] = __builtin_amdgcn_wmma_f32_16x16x32_f16(false, a, false, bf[t],
                                                      (short)0, acc[t], false, false);
  }

  int chbase = (mt << 4) + ((lane & 16) ? 8 : 0);
  float bv[8];
#pragma unroll
  for (int r = 0; r < 8; ++r) bv[r] = bias ? bias[(size_t)b * O + chbase + r] : 0.f;
#pragma unroll
  for (int t = 0; t < 4; ++t) {
    int s = s0 + t * 16 + col;
    float* op = out + ((size_t)b * S + s) * O + chbase;
#pragma unroll
    for (int r = 0; r < 8; ++r) op[r] = acc[t][r] + bv[r];
  }
}

// ---------------------------------------------------------------------------
// Edge assembly: y[b][s=n*K+kk][o] = G[b][idx][o] + H[b][n][o]; contiguous.
// ---------------------------------------------------------------------------
__global__ void edge_assemble_kernel(const float* __restrict__ G, const float* __restrict__ H,
                                     const int* __restrict__ idx, float* __restrict__ y) {
  int t = blockIdx.x * blockDim.x + threadIdx.x;     // B*N*KNN
  if (t >= NBATCH * NPTS * KNN) return;
  int n = (t / KNN) % NPTS;
  int b = t / (KNN * NPTS);
  int j = idx[t];
  const float* Gp = G + ((size_t)b * NPTS + j) * 64;
  const float* Hp = H + ((size_t)b * NPTS + n) * 64;
  float* yp = y + (size_t)t * 64;
#pragma unroll
  for (int o = 0; o < 64; ++o) yp[o] = Gp[o] + Hp[o];
}

// ---------------------------------------------------------------------------
// Per-channel sum/sumsq, point-major input. grid = ((O*SCHUNK)/256, B).
// ---------------------------------------------------------------------------
__global__ void stats_kernel(const float* __restrict__ y, int O, int S, float* __restrict__ stats) {
  int t = blockIdx.x * blockDim.x + threadIdx.x;
  int b = blockIdx.y;
  if (t >= O * SCHUNK) return;
  int c = t % O;
  int ch = t / O;
  int per = S / SCHUNK;
  const float* p = y + ((size_t)b * S + (size_t)ch * per) * O + c;
  float s = 0.f, sq = 0.f;
  for (int i = 0; i < per; ++i) { float v = p[(size_t)i * O]; s += v; sq += v * v; }
  atomicAdd(&stats[c], s);
  atomicAdd(&stats[1024 + c], sq);
}

__global__ void finalize_kernel(const float* __restrict__ stats, const float* __restrict__ g,
                                const float* __restrict__ beta, float cntInv, int C,
                                float* __restrict__ scale, float* __restrict__ shift) {
  int c = blockIdx.x * blockDim.x + threadIdx.x;
  if (c >= C) return;
  float mean = stats[c] * cntInv;
  float var  = stats[1024 + c] * cntInv - mean * mean;
  float inv  = rsqrtf(var + 1e-5f);
  float sc   = g[c] * inv;
  scale[c] = sc;
  shift[c] = beta[c] - mean * sc;
}

// ---------------------------------------------------------------------------
// Fused BN-affine + lrelu, f32 -> f16 (vector x4). total4 = B*S*O/4.
// ---------------------------------------------------------------------------
__global__ void act_f16_kernel(const float* __restrict__ y, const float* __restrict__ scale,
                               const float* __restrict__ shift, half_t* __restrict__ yh,
                               int O, int total4) {
  int t = blockIdx.x * blockDim.x + threadIdx.x;
  if (t >= total4) return;
  size_t i = (size_t)t * 4;
  int c = (int)(i % O);
  v4f x = *(const v4f*)(y + i);
  v4h r;
#pragma unroll
  for (int j = 0; j < 4; ++j) {
    float v = fmaf(scale[c + j], x[j], shift[c + j]);
    r[j] = (half_t)fmaxf(v, 0.2f * v);
  }
  *(v4h*)(yh + i) = r;
}

// x_f32[b][n][o] and x123h[b][n][192] slot = max_k lrelu(scale*z + shift)
__global__ void maxk_kernel(const float* __restrict__ z, const float* __restrict__ scale,
                            const float* __restrict__ shift, float* __restrict__ xo,
                            half_t* __restrict__ x123h, int off) {
  int t = blockIdx.x * blockDim.x + threadIdx.x;     // B*N*64
  if (t >= NBATCH * NPTS * 64) return;
  int o = t & 63;
  int n = (t >> 6) % NPTS;
  int b = t / (64 * NPTS);
  const float* zp = z + (((size_t)b * NPTS + n) * KNN) * 64 + o;
  float sc = scale[o], sh = shift[o];
  float m = -3.4e38f;
#pragma unroll
  for (int kk = 0; kk < KNN; ++kk) {
    float v = fmaf(sc, zp[(size_t)kk * 64], sh);
    m = fmaxf(m, fmaxf(v, 0.2f * v));
  }
  xo[t] = m;
  x123h[((size_t)b * NPTS + n) * 192 + off + o] = (half_t)m;
}

// xg[b][c] = max_n lrelu(scale*y + shift); y point-major [b][n][O]
__global__ void maxn_kernel(const float* __restrict__ y, const float* __restrict__ scale,
                            const float* __restrict__ shift, float* __restrict__ xg, int O) {
  int c = blockIdx.x * blockDim.x + threadIdx.x;
  int b = blockIdx.y;
  if (c >= O) return;
  const float* p = y + (size_t)b * NPTS * O + c;
  float sc = scale[c], sh = shift[c];
  float m = -3.4e38f;
  for (int n = 0; n < NPTS; ++n) {
    float v = fmaf(sc, p[(size_t)n * O], sh);
    m = fmaxf(m, fmaxf(v, 0.2f * v));
  }
  xg[(size_t)b * O + c] = m;
}

// P[b][o] = sum_{c<1024} w7[o,c] * xg[b,c]
__global__ void bias_gemm_kernel(const float* __restrict__ w7, const float* __restrict__ xg,
                                 float* __restrict__ P) {
  int t = blockIdx.x * blockDim.x + threadIdx.x;
  if (t >= NBATCH * 512) return;
  int o = t % 512, b = t / 512;
  float s = 0.f;
  for (int c = 0; c < 1024; ++c) s = fmaf(w7[(size_t)o * 1216 + c], xg[(size_t)b * 1024 + c], s);
  P[t] = s;
}

// out[b][n][o<13] = sum_{c<256} w9[o,c] * lrelu(scale*y8 + shift); y8 [b][n][256]
__global__ void final_kernel(const float* __restrict__ y8, const float* __restrict__ scale,
                             const float* __restrict__ shift, const float* __restrict__ w9,
                             float* __restrict__ out) {
  int t = blockIdx.x * blockDim.x + threadIdx.x;     // B*N
  if (t >= NBATCH * NPTS) return;
  const float* yp = y8 + (size_t)t * 256;
  float acc[13];
#pragma unroll
  for (int o = 0; o < 13; ++o) acc[o] = 0.f;
  for (int c = 0; c < 256; ++c) {
    float v = fmaf(scale[c], yp[c], shift[c]);
    v = fmaxf(v, 0.2f * v);
#pragma unroll
    for (int o = 0; o < 13; ++o) acc[o] = fmaf(w9[(size_t)o * 256 + c], v, acc[o]);
  }
  float* op = out + (size_t)t * 13;
#pragma unroll
  for (int o = 0; o < 13; ++o) op[o] = acc[o];
}

// ---------------------------------------------------------------------------
// Host orchestration
// ---------------------------------------------------------------------------
static void run_pack(const float* W, int ldw, int koff, int Cdiff, int mode,
                     int O, int C, half_t* apack, hipStream_t stm) {
  int total = (O / 16) * (C / 32) * 32;
  pack_a_kernel<<<(total + 255) / 256, 256, 0, stm>>>(W, ldw, koff, Cdiff, mode, O, C, apack);
}

static void run_gemm(const half_t* apack, const half_t* bin, int ldb, int coff,
                     int O, int C, int S, const float* bias, float* out, hipStream_t stm) {
  int waves  = (O / 16) * (S / 64);
  int blocks = (waves + 7) / 8;
  gemm_wmma_kernel<<<dim3(blocks, NBATCH), dim3(256), 0, stm>>>(
      apack, bin, ldb, coff, O, C, S, bias, out);
}

extern "C" void kernel_launch(void* const* d_in, const int* in_sizes, int n_in,
                              void* d_out, int out_size, void* d_ws, size_t ws_size,
                              hipStream_t stream) {
  (void)in_sizes; (void)n_in; (void)out_size; (void)ws_size;
  const float* feat = (const float*)d_in[0];
  const float* w[10];
  for (int i = 1; i <= 9; ++i) w[i] = (const float*)d_in[i];
  const float *g[9], *bb[9];
  for (int i = 1; i <= 8; ++i) { g[i] = (const float*)d_in[10 + 2 * (i - 1)]; bb[i] = (const float*)d_in[11 + 2 * (i - 1)]; }
  float* out = (float*)d_out;

  // workspace layout (float units)
  float* ws  = (float*)d_ws;
  int*   idx = (int*)ws;                               // B*N*K ints
  float* G   = ws + (size_t)NBATCH * NPTS * KNN;
  float* H   = G + (size_t)NBATCH * NPTS * 64;
  float* ya  = H + (size_t)NBATCH * NPTS * 64;         // 84MB f32 (also y6, y8)
  float* yb  = ya + (size_t)NBATCH * SEDGE * 64;       // 84MB f32 (also y7)
  float* x1  = yb + (size_t)NBATCH * SEDGE * 64;       // f32 copies for kNN
  float* x2  = x1 + (size_t)NBATCH * NPTS * 64;
  float* x3  = x2 + (size_t)NBATCH * NPTS * 64;
  float* xg  = x3 + (size_t)NBATCH * NPTS * 64;        // B*1024
  float* P   = xg + (size_t)NBATCH * 1024;             // B*512
  float* st  = P + (size_t)NBATCH * 512;               // 2048 stats
  float* scl = st + 2048;                              // 1024
  float* sft = scl + 1024;                             // 1024
  half_t* apack  = (half_t*)(sft + 1024);              // up to 1024*192 halfs
  half_t* yah    = (half_t*)(sft + 1024 + 98304);      // B*SEDGE*64 halfs (also y7h)
  half_t* x123h  = yah + (size_t)NBATCH * SEDGE * 64;  // B*N*192 halfs

  const float cntE = 1.f / (float)(NBATCH * NPTS * KNN);
  const float cntD = 1.f / (float)(NBATCH * NPTS);
  const int TPB = 256;
  int gBN  = (NBATCH * NPTS + TPB - 1) / TPB;
  int gBNK = (NBATCH * NPTS * KNN + TPB - 1) / TPB;
  int gX   = (NBATCH * NPTS * 64 + TPB - 1) / TPB;
  int tE4  = NBATCH * SEDGE * 64 / 4;                  // act elements/4, edge tensors
  int t74  = NBATCH * NPTS * 512 / 4;                  // act elements/4, y7

#define ZSTATS() hipMemsetAsync(st, 0, 2048 * sizeof(float), stream)
#define STATS(buf, O_, S_) stats_kernel<<<dim3(((O_) * SCHUNK + TPB - 1) / TPB, NBATCH), TPB, 0, stream>>>(buf, O_, S_, st)

  // ---------------- stage 1 (blocks 1,2; coords = channels 6..7) ------------
  knn2_kernel<<<gBN, TPB, 0, stream>>>(feat, idx);
  gh8_kernel<<<gBN, TPB, 0, stream>>>(w[1], feat, G, H);
  edge_assemble_kernel<<<gBNK, TPB, 0, stream>>>(G, H, idx, ya);
  ZSTATS(); STATS(ya, 64, SEDGE);
  finalize_kernel<<<1, TPB, 0, stream>>>(st, g[1], bb[1], cntE, 64, scl, sft);
  act_f16_kernel<<<(tE4 + TPB - 1) / TPB, TPB, 0, stream>>>(ya, scl, sft, yah, 64, tE4);
  run_pack(w[2], 64, 0, 0, 0, 64, 64, apack, stream);
  run_gemm(apack, yah, 64, 0, 64, 64, SEDGE, nullptr, yb, stream);
  ZSTATS(); STATS(yb, 64, SEDGE);
  finalize_kernel<<<1, TPB, 0, stream>>>(st, g[2], bb[2], cntE, 64, scl, sft);
  maxk_kernel<<<gX, TPB, 0, stream>>>(yb, scl, sft, x1, x123h, 0);

  // ---------------- stage 2 (blocks 3,4) ------------------------------------
  knn64_kernel<<<gBN, TPB, 0, stream>>>(x1, idx);
  run_pack(w[3], 128, 0, 0, 0, 64, 64, apack, stream);
  run_gemm(apack, x123h, 192, 0, 64, 64, NPTS, nullptr, G, stream);
  run_pack(w[3], 128, 0, 64, 1, 64, 64, apack, stream);
  run_gemm(apack, x123h, 192, 0, 64, 64, NPTS, nullptr, H, stream);
  edge_assemble_kernel<<<gBNK, TPB, 0, stream>>>(G, H, idx, ya);
  ZSTATS(); STATS(ya, 64, SEDGE);
  finalize_kernel<<<1, TPB, 0, stream>>>(st, g[3], bb[3], cntE, 64, scl, sft);
  act_f16_kernel<<<(tE4 + TPB - 1) / TPB, TPB, 0, stream>>>(ya, scl, sft, yah, 64, tE4);
  run_pack(w[4], 64, 0, 0, 0, 64, 64, apack, stream);
  run_gemm(apack, yah, 64, 0, 64, 64, SEDGE, nullptr, yb, stream);
  ZSTATS(); STATS(yb, 64, SEDGE);
  finalize_kernel<<<1, TPB, 0, stream>>>(st, g[4], bb[4], cntE, 64, scl, sft);
  maxk_kernel<<<gX, TPB, 0, stream>>>(yb, scl, sft, x2, x123h, 64);

  // ---------------- stage 3 (block 5) ---------------------------------------
  knn64_kernel<<<gBN, TPB, 0, stream>>>(x2, idx);
  run_pack(w[5], 128, 0, 0, 0, 64, 64, apack, stream);
  run_gemm(apack, x123h, 192, 64, 64, 64, NPTS, nullptr, G, stream);
  run_pack(w[5], 128, 0, 64, 1, 64, 64, apack, stream);
  run_gemm(apack, x123h, 192, 64, 64, 64, NPTS, nullptr, H, stream);
  edge_assemble_kernel<<<gBNK, TPB, 0, stream>>>(G, H, idx, ya);
  ZSTATS(); STATS(ya, 64, SEDGE);
  finalize_kernel<<<1, TPB, 0, stream>>>(st, g[5], bb[5], cntE, 64, scl, sft);
  maxk_kernel<<<gX, TPB, 0, stream>>>(ya, scl, sft, x3, x123h, 128);

  // ---------------- block 6: 1024 x 192, then global max -> xg --------------
  float* y6 = ya;
  run_pack(w[6], 192, 0, 0, 0, 1024, 192, apack, stream);
  run_gemm(apack, x123h, 192, 0, 1024, 192, NPTS, nullptr, y6, stream);
  ZSTATS(); STATS(y6, 1024, NPTS);
  finalize_kernel<<<4, TPB, 0, stream>>>(st, g[6], bb[6], cntD, 1024, scl, sft);
  maxn_kernel<<<dim3(4, NBATCH), TPB, 0, stream>>>(y6, scl, sft, xg, 1024);

  // ---------------- block 7: 512 x 1216 = bias(xg part) + 192-ch GEMM -------
  float* y7 = yb;
  bias_gemm_kernel<<<(NBATCH * 512 + TPB - 1) / TPB, TPB, 0, stream>>>(w[7], xg, P);
  run_pack(w[7], 1216, 1024, 0, 0, 512, 192, apack, stream);
  run_gemm(apack, x123h, 192, 0, 512, 192, NPTS, P, y7, stream);
  ZSTATS(); STATS(y7, 512, NPTS);
  finalize_kernel<<<2, TPB, 0, stream>>>(st, g[7], bb[7], cntD, 512, scl, sft);

  // ---------------- block 8: 256 x 512 --------------------------------------
  float* y8 = ya;
  half_t* y7h = yah;                                   // reuse edge f16 buffer
  act_f16_kernel<<<(t74 + TPB - 1) / TPB, TPB, 0, stream>>>(y7, scl, sft, y7h, 512, t74);
  run_pack(w[8], 512, 0, 0, 0, 256, 512, apack, stream);
  run_gemm(apack, y7h, 512, 0, 256, 512, NPTS, nullptr, y8, stream);
  ZSTATS(); STATS(y8, 256, NPTS);
  finalize_kernel<<<1, TPB, 0, stream>>>(st, g[8], bb[8], cntD, 256, scl, sft);

  // ---------------- block 9: 13 x 256 -> output (B, N, 13) ------------------
  final_kernel<<<gBN, TPB, 0, stream>>>(y8, scl, sft, w[9], out);

#undef ZSTATS
#undef STATS
}
